// CA_SHSA_42752104464792
// MI455X (gfx1250) — matrix operand order repaired and checked by reading, not compiled, not run on hardware
//
#include <hip/hip_runtime.h>
#include <hip/hip_bf16.h>
#include <stdint.h>

#define BB 16
#define CC 512
#define NN 4096
#define TILE 128
#define KSTEP 32
#define LDS_STRIDE 40   // 32 elems + 8 pad -> 80B row stride, keeps 16B alignment for b128

typedef __attribute__((ext_vector_type(16))) __bf16 v16bf;
typedef __attribute__((ext_vector_type(8)))  float  v8f;
typedef __attribute__((ext_vector_type(4)))  float  vf4;
typedef int v4i __attribute__((vector_size(16)));

union frag_u { uint4 u[2]; v16bf v; __bf16 e[16]; };

#if defined(__gfx1250__) && __has_builtin(__builtin_amdgcn_global_load_async_to_lds_b128)
#define HAVE_ASYNC_LDS 1
typedef v4i __attribute__((address_space(1)))* gptr4;
typedef v4i __attribute__((address_space(3)))* lptr4;
#else
#define HAVE_ASYNC_LDS 0
#endif

// copy 16 bytes (8 bf16) global -> LDS; async DMA on CDNA5, sync fallback otherwise
__device__ __forceinline__ void cp16(const __bf16* g, __bf16* l) {
#if HAVE_ASYNC_LDS
    gptr4 gp = (gptr4)(unsigned long long)(uintptr_t)g;        // flat -> global (same address)
    lptr4 lp = (lptr4)(unsigned int)(uintptr_t)l;              // flat -> LDS (low 32 bits)
    __builtin_amdgcn_global_load_async_to_lds_b128(gp, lp, 0, 0);
#else
    *(uint4*)l = *(const uint4*)g;
#endif
}

template <int N>
__device__ __forceinline__ void wait_async() {
#if HAVE_ASYNC_LDS
#if __has_builtin(__builtin_amdgcn_s_wait_asynccnt)
    __builtin_amdgcn_s_wait_asynccnt(N);
#else
    asm volatile("s_wait_asynccnt %0" :: "n"(N) : "memory");
#endif
#endif
}

// ---------------- kernel 0: fp32 -> bf16 convert ----------------
__global__ __launch_bounds__(256)
void cvt_kernel(const float* __restrict__ x, __bf16* __restrict__ Xb)
{
    size_t i = ((size_t)blockIdx.x * 256 + threadIdx.x) * 8;
    vf4 f0 = __builtin_nontemporal_load((const vf4*)(x + i));
    vf4 f1 = __builtin_nontemporal_load((const vf4*)(x + i + 4));
    union { __bf16 h[8]; uint4 u; } pk;
    pk.h[0] = (__bf16)f0.x; pk.h[1] = (__bf16)f0.y;
    pk.h[2] = (__bf16)f0.z; pk.h[3] = (__bf16)f0.w;
    pk.h[4] = (__bf16)f1.x; pk.h[5] = (__bf16)f1.y;
    pk.h[6] = (__bf16)f1.z; pk.h[7] = (__bf16)f1.w;
    *(uint4*)(Xb + i) = pk.u;
}

// ---------------- kernel 1: Gram matrix S = Xb Xb^T, scaled logits ----------------
// grid: (16 tiles, 16 batches), block 256 = 8 waves.
// 128x128 tile per block; each wave: 2 (M) x 4 (N) 16x16 WMMA tiles.
// Double-buffered LDS, async global->LDS staging of next K-step while computing.
__global__ __launch_bounds__(256)
void gram_kernel(const __bf16* __restrict__ Xb, const float* __restrict__ wq,
                 const float* __restrict__ wk, float* __restrict__ logits)
{
    __shared__ __bf16 ldsA[2][TILE * LDS_STRIDE];
    __shared__ __bf16 ldsB[2][TILE * LDS_STRIDE];

    const int tid   = threadIdx.x;
    const int lane  = tid & 31;
    const int wave  = tid >> 5;
    const int b     = blockIdx.y;
    const int cBase = (blockIdx.x & 3) * TILE;
    const int dBase = (blockIdx.x >> 2) * TILE;

    const __bf16* Xbb = Xb + (size_t)b * CC * NN;

    const v8f vzero = {0.f,0.f,0.f,0.f,0.f,0.f,0.f,0.f};
    v8f acc[2][4];
    #pragma unroll
    for (int i = 0; i < 2; ++i)
        #pragma unroll
        for (int j = 0; j < 4; ++j) acc[i][j] = vzero;

    const int rowL  = tid >> 1;         // 0..127 staged row
    const int chk   = (tid & 1) * 16;   // elem offset within row (0 or 16)
    const int mt0   = (wave & 3) * 2;
    const int nt0   = (wave >> 2) * 4;
    const int lhalf = lane >> 4;
    const int lmod  = lane & 15;
    const int koffA = lhalf * 8;        // A frag: chunks {b, b+16}, b = 0 or 8
    const int koffB = lhalf * 8;        // B frag: chunks {b, b+16}, b = 0 or 8

    const __bf16* gA0 = Xbb + (size_t)(cBase + rowL) * NN + chk;
    const __bf16* gB0 = Xbb + (size_t)(dBase + rowL) * NN + chk;
    const int ldsOff  = rowL * LDS_STRIDE + chk;

    // prologue: stage k0 = 0 into buffer 0 (4 async instructions / wave)
    cp16(gA0,     &ldsA[0][ldsOff]);
    cp16(gA0 + 8, &ldsA[0][ldsOff + 8]);
    cp16(gB0,     &ldsB[0][ldsOff]);
    cp16(gB0 + 8, &ldsB[0][ldsOff + 8]);

    int cur = 0;
    for (int k0 = 0; k0 < NN; k0 += KSTEP) {
        int knext = k0 + KSTEP;
        if (knext >= NN) knext = 0;     // dummy wrap prefetch on last iter (keeps wait imm constant)
        const int nb = cur ^ 1;
        cp16(gA0 + knext,     &ldsA[nb][ldsOff]);
        cp16(gA0 + knext + 8, &ldsA[nb][ldsOff + 8]);
        cp16(gB0 + knext,     &ldsB[nb][ldsOff]);
        cp16(gB0 + knext + 8, &ldsB[nb][ldsOff + 8]);
        wait_async<4>();                // drain previous buffer's 4, keep the new 4 in flight
        __syncthreads();

        const __bf16* la = ldsA[cur];
        const __bf16* lb = ldsB[cur];
        v16bf af[2], bfr[4];
        #pragma unroll
        for (int i = 0; i < 2; ++i) {
            const __bf16* p = &la[((mt0 + i) * 16 + lmod) * LDS_STRIDE + koffA];
            frag_u f; f.u[0] = *(const uint4*)p; f.u[1] = *(const uint4*)(p + 16);
            af[i] = f.v;
        }
        #pragma unroll
        for (int j = 0; j < 4; ++j) {
            const __bf16* p = &lb[((nt0 + j) * 16 + lmod) * LDS_STRIDE + koffB];
            frag_u f; f.u[0] = *(const uint4*)p; f.u[1] = *(const uint4*)(p + 16);
            bfr[j] = f.v;
        }
        #pragma unroll
        for (int i = 0; i < 2; ++i)
            #pragma unroll
            for (int j = 0; j < 4; ++j)
                acc[i][j] = __builtin_amdgcn_wmma_f32_16x16x32_bf16(
                    false, af[i], false, bfr[j], (short)0, acc[i][j], false, false);
        __syncthreads();
        cur ^= 1;
    }

    // logits = acc * wq[c] * wk[d] / sqrt(N)
    float* Lb = logits + (size_t)b * CC * CC;
    const float scl = 0.015625f; // 1/64
    #pragma unroll
    for (int i = 0; i < 2; ++i) {
        #pragma unroll
        for (int j = 0; j < 4; ++j) {
            const int d = dBase + (nt0 + j) * 16 + lmod;
            const float wkd = wk[d] * scl;
            #pragma unroll
            for (int r = 0; r < 8; ++r) {
                const int c = cBase + (mt0 + i) * 16 + r + 8 * lhalf;
                Lb[(size_t)c * CC + d] = acc[i][j][r] * wq[c] * wkd;
            }
        }
    }
}

// ---------------- kernel 2: row softmax, fold wv, emit bf16 P ----------------
__global__ __launch_bounds__(256)
void softmax_kernel(const float* __restrict__ logits, const float* __restrict__ wv,
                    __bf16* __restrict__ P)
{
    __shared__ float red[256];
    const int row = blockIdx.x;      // b*C + c
    const int tid = threadIdx.x;
    const float* L = logits + (size_t)row * CC;
    float l0 = L[tid], l1 = L[tid + 256];

    red[tid] = fmaxf(l0, l1);
    __syncthreads();
    for (int s = 128; s > 0; s >>= 1) {
        if (tid < s) red[tid] = fmaxf(red[tid], red[tid + s]);
        __syncthreads();
    }
    const float M = red[0];
    __syncthreads();

    float e0 = __expf(l0 - M), e1 = __expf(l1 - M);
    red[tid] = e0 + e1;
    __syncthreads();
    for (int s = 128; s > 0; s >>= 1) {
        if (tid < s) red[tid] += red[tid + s];
        __syncthreads();
    }
    const float inv = 1.f / red[0];

    __bf16* Pr = P + (size_t)row * CC;
    Pr[tid]       = (__bf16)(e0 * inv * wv[tid]);
    Pr[tid + 256] = (__bf16)(e1 * inv * wv[tid + 256]);
}

// ---------------- kernel 3: out = P @ Xb (fp32 out) ----------------
// grid: (4 Mtiles * 32 Ntiles = 128, 16 batches)
// A tile (P) staged with async DMA; B tile (Xb) needs a transpose -> register path.
__global__ __launch_bounds__(256)
void out_kernel(const __bf16* __restrict__ P, const __bf16* __restrict__ Xb,
                float* __restrict__ out)
{
    __shared__ __bf16 ldsA[2][TILE * LDS_STRIDE];
    __shared__ __bf16 ldsB[2][TILE * LDS_STRIDE]; // transposed: [n_local][k]

    const int tid   = threadIdx.x;
    const int lane  = tid & 31;
    const int wave  = tid >> 5;
    const int b     = blockIdx.y;
    const int cBase = (blockIdx.x & 3) * TILE;
    const int nBase = (blockIdx.x >> 2) * TILE;

    const __bf16* Pb  = P  + (size_t)b * CC * CC;
    const __bf16* Xbb = Xb + (size_t)b * CC * NN;

    const v8f vzero = {0.f,0.f,0.f,0.f,0.f,0.f,0.f,0.f};
    v8f acc[2][4];
    #pragma unroll
    for (int i = 0; i < 2; ++i)
        #pragma unroll
        for (int j = 0; j < 4; ++j) acc[i][j] = vzero;

    const int rowL  = tid >> 1;
    const int chk   = (tid & 1) * 16;
    const int krow  = tid >> 3;        // 0..31 : staged k row (B tile)
    const int ncol  = (tid & 7) * 16;  // 0..112: first n column of 16-elem chunk
    const int mt0   = (wave & 3) * 2;
    const int nt0   = (wave >> 2) * 4;
    const int lhalf = lane >> 4;
    const int lmod  = lane & 15;
    const int koffA = lhalf * 8;
    const int koffB = lhalf * 8;

    const __bf16* gA0 = Pb + (size_t)(cBase + rowL) * CC + chk;
    const __bf16* gB0 = Xbb + (size_t)krow * NN + nBase + ncol;
    const int ldsOffA = rowL * LDS_STRIDE + chk;

    // B tile transpose stage (sync; DMA cannot transpose)
    auto stageB = [&](int kk, int buf) {
        frag_u fb;
        fb.u[0] = *(const uint4*)(gB0 + (size_t)kk * NN);
        fb.u[1] = *(const uint4*)(gB0 + (size_t)kk * NN + 8);
        #pragma unroll
        for (int j = 0; j < 16; ++j)
            ldsB[buf][(ncol + j) * LDS_STRIDE + krow] = fb.e[j];
    };

    // prologue: stage k0 = 0 into buffer 0 (A: 2 async instrs / wave)
    cp16(gA0,     &ldsA[0][ldsOffA]);
    cp16(gA0 + 8, &ldsA[0][ldsOffA + 8]);
    stageB(0, 0);

    int cur = 0;
    for (int k0 = 0; k0 < CC; k0 += KSTEP) {
        int knext = k0 + KSTEP;
        if (knext >= CC) knext = 0;     // dummy wrap prefetch on last iter
        const int nb = cur ^ 1;
        cp16(gA0 + knext,     &ldsA[nb][ldsOffA]);
        cp16(gA0 + knext + 8, &ldsA[nb][ldsOffA + 8]);
        stageB(knext, nb);
        wait_async<2>();
        __syncthreads();

        const __bf16* la = ldsA[cur];
        const __bf16* lb = ldsB[cur];
        v16bf af[2], bfr[4];
        #pragma unroll
        for (int i = 0; i < 2; ++i) {
            const __bf16* p = &la[((mt0 + i) * 16 + lmod) * LDS_STRIDE + koffA];
            frag_u f; f.u[0] = *(const uint4*)p; f.u[1] = *(const uint4*)(p + 16);
            af[i] = f.v;
        }
        #pragma unroll
        for (int j = 0; j < 4; ++j) {
            const __bf16* p = &lb[((nt0 + j) * 16 + lmod) * LDS_STRIDE + koffB];
            frag_u f; f.u[0] = *(const uint4*)p; f.u[1] = *(const uint4*)(p + 16);
            bfr[j] = f.v;
        }
        #pragma unroll
        for (int i = 0; i < 2; ++i)
            #pragma unroll
            for (int j = 0; j < 4; ++j)
                acc[i][j] = __builtin_amdgcn_wmma_f32_16x16x32_bf16(
                    false, af[i], false, bfr[j], (short)0, acc[i][j], false, false);
        __syncthreads();
        cur ^= 1;
    }

    float* Ob = out + (size_t)b * CC * NN;
    #pragma unroll
    for (int i = 0; i < 2; ++i) {
        #pragma unroll
        for (int j = 0; j < 4; ++j) {
            const int n = nBase + (nt0 + j) * 16 + lmod;
            #pragma unroll
            for (int r = 0; r < 8; ++r) {
                const int c = cBase + (mt0 + i) * 16 + r + 8 * lhalf;
                __builtin_nontemporal_store(acc[i][j][r], &Ob[(size_t)c * NN + n]);
            }
        }
    }
}

extern "C" void kernel_launch(void* const* d_in, const int* in_sizes, int n_in,
                              void* d_out, int out_size, void* d_ws, size_t ws_size,
                              hipStream_t stream) {
    const float* x  = (const float*)d_in[0];
    const float* wq = (const float*)d_in[1];
    const float* wk = (const float*)d_in[2];
    const float* wv = (const float*)d_in[3];
    float* out = (float*)d_out;

    char* ws = (char*)d_ws;
    __bf16* Xb     = (__bf16*)ws;                           // 64 MiB bf16 X
    float*  logits = (float*)(ws + ((size_t)64 << 20));     // 16 MiB fp32 logits
    __bf16* P      = (__bf16*)(ws + ((size_t)80 << 20));    //  8 MiB bf16 P

    cvt_kernel<<<dim3(16384), dim3(256), 0, stream>>>(x, Xb);
    gram_kernel<<<dim3(16, BB), dim3(256), 0, stream>>>(Xb, wq, wk, logits);
    softmax_kernel<<<dim3(BB * CC), dim3(256), 0, stream>>>(logits, wv, P);
    out_kernel<<<dim3(128, BB), dim3(256), 0, stream>>>(P, Xb, out);
}
